// Recurrence_146028888239
// MI455X (gfx1250) — matrix operand order repaired and checked by reading, not compiled, not run on hardware
//
#include <hip/hip_runtime.h>
#include <hip/hip_bf16.h>

typedef __attribute__((ext_vector_type(16))) _Float16 v16h;
typedef __attribute__((ext_vector_type(8)))  _Float16 v8h;
typedef __attribute__((ext_vector_type(8)))  float    v8f;

#define T_   32
#define N_   64
#define E_   256
#define HID_ 1024
#define A_   32
#define S_   32
#define HP_  8
#define INF_ 1e8f

// d_out layout (floats), reference tuple order
#define SL_OFF  0        // search_logits (S,N,A)   = 65536
#define PA_OFF  65536    // planned_actions (N,HP)  = 512
#define XO_OFF  66048    // X (N,HP,E)              = 131072
#define VAL_OFF 197120   // value (T,N)             = 2048
#define MO_OFF  199168   // model_loss (T,N,E)      = 524288
#define EL_OFF  723456   // embed_loss (T,N)        = 2048

// ---------------------------------------------------------------------------
// Generic WMMA GEMM:  out[M,N] = epi( A[M,K](f16) @ B[N,K](f16)^T + bias[N] )
// One wave per 16(M) x 64(N) strip: 4 accumulators reuse one A fragment.
// N must be a multiple of 16 (>= 16); overhanging tiles are clamped to the
// last valid tile (redundant but branch-free -> unconditional WMMA, no
// accumulator copies). A and B fragments use contiguous k order, k = 16*hi + s (wave32).
// ---------------------------------------------------------------------------
__device__ __forceinline__ v8f wmma_f16(v16h a, v16h b, v8f c) {
  return __builtin_amdgcn_wmma_f32_16x16x32_f16(false, a, false, b, (short)0, c,
                                                false, false);
}

__device__ __forceinline__ v16h ldfrag16(const _Float16* p0, const _Float16* p1) {
  v8h a0 = *(const v8h*)p0;
  v8h a1 = *(const v8h*)p1;
  return __builtin_shufflevector(a0, a1,
      0,1,2,3,4,5,6,7,8,9,10,11,12,13,14,15);
}

__global__ void __launch_bounds__(32) k_gemm(
    const _Float16* __restrict__ A, const _Float16* __restrict__ B,
    const float* __restrict__ bias, int M, int N, int K, int ldo,
    float* __restrict__ outF, _Float16* __restrict__ outH,
    const float* __restrict__ sub, int relu)
{
  const int lane = threadIdx.x;
  const int hi   = lane >> 4;          // half-wave select
  const int l16  = lane & 15;
  const int m0   = blockIdx.y << 4;
  const int nblk = blockIdx.x << 6;
  if (m0 >= M) return;

  // clamp tiles into range (branch-free inner loop)
  int n0s[4];
  const _Float16* brow[4];
  #pragma unroll
  for (int t = 0; t < 4; ++t) {
    int n0 = nblk + (t << 4);
    if (n0 > N - 16) n0 = N - 16;
    n0s[t] = n0;
    // B^T 32x16 fragment base: lanes 0-15 K=k0..k0+15, lanes 16-31 +16
    brow[t] = B + (size_t)(n0 + l16) * K + (hi ? 16 : 0);
  }
  // A 16x32 fragment base: lane hi=0 covers K {0..15}, hi=1 {16..31}
  const _Float16* arow = A + (size_t)(m0 + l16) * K + (hi ? 16 : 0);

  v8f acc0 = {}, acc1 = {}, acc2 = {}, acc3 = {};
  for (int k0 = 0; k0 < K; k0 += 32) {
    v16h af  = ldfrag16(arow + k0,    arow + k0 + 8);
    v16h bf0 = ldfrag16(brow[0] + k0, brow[0] + k0 + 8);
    v16h bf1 = ldfrag16(brow[1] + k0, brow[1] + k0 + 8);
    v16h bf2 = ldfrag16(brow[2] + k0, brow[2] + k0 + 8);
    v16h bf3 = ldfrag16(brow[3] + k0, brow[3] + k0 + 8);
    acc0 = wmma_f16(af, bf0, acc0);
    acc1 = wmma_f16(af, bf1, acc1);
    acc2 = wmma_f16(af, bf2, acc2);
    acc3 = wmma_f16(af, bf3, acc3);
  }

  // Epilogue: D vgpr r -> row m0 + r + hi*8, col n0 + l16
  v8f accs[4] = { acc0, acc1, acc2, acc3 };
  #pragma unroll
  for (int t = 0; t < 4; ++t) {
    int n = n0s[t] + l16;
    float bv = bias ? bias[n] : 0.0f;
    #pragma unroll
    for (int r = 0; r < 8; ++r) {
      int m = m0 + (hi ? 8 : 0) + r;
      size_t idx = (size_t)m * ldo + n;
      float v = accs[t][r] + bv;
      if (relu) v = fmaxf(v, 0.0f);
      if (sub) { float d = v - sub[idx]; v = d * d; }   // model-loss epilogue
      if (outF) outF[idx] = v;
      if (outH) outH[idx] = (_Float16)v;
    }
  }
}

// ---------------------------------------------------------------------------
// small helper kernels
// ---------------------------------------------------------------------------
__global__ void k_convert(const float* __restrict__ s, _Float16* __restrict__ d, int n) {
  int i = blockIdx.x * blockDim.x + threadIdx.x;
  if (i < n) d[i] = (_Float16)s[i];
}

// emb[t,n,e] = embed1[obs[t,n,e/16], e%16]; also f16 copy of the t=0 rows
__global__ void k_embed(const int* __restrict__ obs, const float* __restrict__ embed1,
                        float* __restrict__ emb, _Float16* __restrict__ embh) {
  int i = blockIdx.x * blockDim.x + threadIdx.x;           // < 2048*256
  if (i >= T_ * N_ * E_) return;
  int row = i >> 8, e = i & 255;
  float v = embed1[obs[row * 16 + (e >> 4)] * 16 + (e & 15)];
  emb[i] = v;
  if (row < N_) embh[i] = (_Float16)v;                     // t == 0 slab
}

__global__ void k_init(float* __restrict__ X, float* __restrict__ lg,
                       int* __restrict__ I, int* __restrict__ pa) {
  int i = blockIdx.x * blockDim.x + threadIdx.x;
  if (i < N_ * HP_ * E_) X[i] = 0.0f;
  if (i < N_ * A_)       lg[i] = 0.0f;
  if (i < N_ * HP_)      pa[i] = 0;
  if (i < N_)            I[i]  = 0;
}

// GRU step with h0 = 0 (gh == bhh):  h = (1-z) * tanh(inn + r*hn)
__global__ void k_gru0(const float* __restrict__ gi, const float* __restrict__ bhh,
                       _Float16* __restrict__ h0h) {
  int i = blockIdx.x * blockDim.x + threadIdx.x;           // < 64*1024
  if (i >= N_ * HID_) return;
  int n = i >> 10, j = i & 1023;
  const float* g = gi + (size_t)n * 3 * HID_;
  float r  = 1.0f / (1.0f + expf(-(g[j]            + bhh[j])));
  float z  = 1.0f / (1.0f + expf(-(g[HID_ + j]     + bhh[HID_ + j])));
  float nn = tanhf(g[2 * HID_ + j] + r * bhh[2 * HID_ + j]);
  h0h[i] = (_Float16)((1.0f - z) * nn);
}

// x = X[n, I[n]]; xa = relu(x)  (f32 + f16 copies)
__global__ void k_gather(const float* __restrict__ X, const int* __restrict__ I,
                         float* __restrict__ xcur, float* __restrict__ xaf,
                         _Float16* __restrict__ xah) {
  int i = blockIdx.x * blockDim.x + threadIdx.x;           // < 64*256
  if (i >= N_ * E_) return;
  int n = i >> 8, e = i & 255;
  float x = X[((size_t)n * HP_ + I[n]) * E_ + e];
  xcur[i] = x;
  float a = fmaxf(x, 0.0f);
  xaf[i] = a;
  xah[i] = (_Float16)a;
}

// flat[n, a*E+e] = relu(embed_opt[Ps[n], a] * x[n, e])   (plan step)
__global__ void k_flat_plan(const float* __restrict__ xcur, const float* __restrict__ eopt,
                            const int* __restrict__ Ps, _Float16* __restrict__ flat) {
  int i = blockIdx.x * blockDim.x + threadIdx.x;           // < 64*8192
  if (i >= N_ * A_ * E_) return;
  int n = i >> 13, rem = i & 8191, ai = rem >> 8, e = rem & 255;
  float v = eopt[Ps[n] * A_ + ai] * xcur[n * E_ + e];
  flat[i] = (_Float16)fmaxf(v, 0.0f);
}

// time-loop flat for chunk c (8 timesteps = 512 rows), prev with wrap at t=0
__global__ void k_flat_time(const float* __restrict__ emb, const float* __restrict__ eopt,
                            const int* __restrict__ P, _Float16* __restrict__ flat, int c) {
  int i = blockIdx.x * blockDim.x + threadIdx.x;           // < 512*8192
  if (i >= 512 * A_ * E_) return;
  int row = i >> 13, rem = i & 8191, ai = rem >> 8, e = rem & 255;
  int t = c * 8 + (row >> 6), n = row & 63;
  int tp = (t + T_ - 1) & (T_ - 1);
  float x  = emb[((size_t)tp * N_ + n) * E_ + e];
  float ao = eopt[P[t * N_ + n] * A_ + ai];
  flat[i] = (_Float16)fmaxf(ao * x, 0.0f);
}

// per-plan-step state update: sharp, logits, stack push/pop, scatter pa
__global__ void __launch_bounds__(64) k_update(
    const int* __restrict__ Ps, const float* __restrict__ xaf,
    const float* __restrict__ Ws, const float* __restrict__ bs,
    const float* __restrict__ vals, const float* __restrict__ xnext,
    float* __restrict__ logits, int* __restrict__ I, int* __restrict__ pa,
    float* __restrict__ X, float* __restrict__ slout)
{
  int n = threadIdx.x;
  if (n >= N_) return;
  int a = Ps[n];
  float sh = bs[0];
  const float* xr = xaf + n * E_;
  for (int e = 0; e < E_; ++e) sh += xr[e] * Ws[e];
  float* lg = logits + n * A_;
  bool isnew = true;
  for (int j = 0; j < A_; ++j) isnew = isnew && (lg[j] == 0.0f);
  const float* vr = vals + n * A_;
  float* so = slout + n * A_;
  for (int j = 0; j < A_; ++j) {
    float v = isnew ? sh * vr[j] : lg[j];
    so[j] = v;
    lg[j] = v - (j == a ? INF_ : 0.0f);
  }
  bool push = vr[a] > 0.0f;
  int Io = I[n];
  pa[n * HP_ + Io] = a;                                   // scatter at old I
  int Iu = Io + 1; if (Iu > HP_ - 1) Iu = HP_ - 1;
  if (push) {
    const float* xn = xnext + n * E_;
    float* xd = X + ((size_t)n * HP_ + Iu) * E_;
    for (int e = 0; e < E_; ++e) xd[e] = xn[e];
  }
  I[n] = push ? Iu : (Io > 0 ? Io - 1 : 0);
}

// embed_loss: -entropy of softmax(sharp * cos(emb, X[n,h])) over HP
__global__ void __launch_bounds__(32) k_embed_loss(
    const float* __restrict__ emb, const float* __restrict__ X,
    const float* __restrict__ Ws, const float* __restrict__ bs,
    float* __restrict__ out)
{
  int row = blockIdx.x;                                   // t*64+n, < 2048
  int n = row & 63, lane = threadIdx.x;
  const float* er = emb + (size_t)row * E_ + lane * 8;
  float ev[8], se = 0.0f, sp = 0.0f;
  #pragma unroll
  for (int j = 0; j < 8; ++j) {
    ev[j] = er[j];
    se += ev[j] * ev[j];
    sp += fmaxf(ev[j], 0.0f) * Ws[lane * 8 + j];
  }
  for (int o = 16; o; o >>= 1) { se += __shfl_xor(se, o, 32); sp += __shfl_xor(sp, o, 32); }
  float ne = sqrtf(se) + 1e-8f;
  float sharp = sp + bs[0];
  float lg[HP_];
  for (int h = 0; h < HP_; ++h) {
    const float* xr = X + ((size_t)n * HP_ + h) * E_ + lane * 8;
    float d = 0.0f, sx = 0.0f;
    #pragma unroll
    for (int j = 0; j < 8; ++j) { float xv = xr[j]; d += ev[j] * xv; sx += xv * xv; }
    for (int o = 16; o; o >>= 1) { d += __shfl_xor(d, o, 32); sx += __shfl_xor(sx, o, 32); }
    lg[h] = sharp * (d / (ne * (sqrtf(sx) + 1e-8f)));
  }
  float m = lg[0];
  for (int h = 1; h < HP_; ++h) m = fmaxf(m, lg[h]);
  float s = 0.0f;
  for (int h = 0; h < HP_; ++h) s += expf(lg[h] - m);
  float lse = m + logf(s);
  float el = 0.0f;
  for (int h = 0; h < HP_; ++h) { float lp = lg[h] - lse; el += expf(lp) * lp; }
  if (lane == 0) out[row] = el;
}

// value gather + planned_actions + X copy into d_out
__global__ void k_finalize(const float* __restrict__ X, const int* __restrict__ pa,
                           const int* __restrict__ P, float* __restrict__ out) {
  int i = blockIdx.x * blockDim.x + threadIdx.x;
  if (i < N_ * HP_ * E_) out[XO_OFF + i] = X[i];
  if (i < T_ * N_)       out[VAL_OFF + i] = out[SL_OFF + i * A_ + P[i]];
  if (i < N_ * HP_)      out[PA_OFF + i] = (float)pa[i];
}

// ---------------------------------------------------------------------------
extern "C" void kernel_launch(void* const* d_in, const int* in_sizes, int n_in,
                              void* d_out, int out_size, void* d_ws, size_t ws_size,
                              hipStream_t stream) {
  (void)in_sizes; (void)n_in; (void)out_size; (void)ws_size;
  const int*   obs  = (const int*)  d_in[0];
  const int*   P    = (const int*)  d_in[1];
  const float* emb1 = (const float*)d_in[2];
  const float* Wih  = (const float*)d_in[3];
  /* d_in[4] = Whh is dead: h0 == 0  =>  gh == bhh */
  const float* bih  = (const float*)d_in[5];
  const float* bhh  = (const float*)d_in[6];
  const float* We2  = (const float*)d_in[7];
  const float* be2  = (const float*)d_in[8];
  const float* Ws   = (const float*)d_in[9];
  const float* bs   = (const float*)d_in[10];
  const float* Wc   = (const float*)d_in[11];
  const float* bc   = (const float*)d_in[12];
  const float* Wm1  = (const float*)d_in[13];
  const float* bm1  = (const float*)d_in[14];
  const float* Wm2  = (const float*)d_in[15];
  const float* bm2  = (const float*)d_in[16];
  const float* eopt = (const float*)d_in[17];
  float* out = (float*)d_out;

  char* w = (char*)d_ws;
  size_t off = 0;
  auto alloc = [&](size_t bytes) -> void* {
    void* p = w + off;
    off = (off + bytes + 255) & ~(size_t)255;
    return p;
  };
  float*    emb    = (float*)   alloc((size_t)T_*N_*E_*4);
  _Float16* embh   = (_Float16*)alloc((size_t)N_*E_*2);
  _Float16* WihH   = (_Float16*)alloc((size_t)3*HID_*E_*2);
  _Float16* We2H   = (_Float16*)alloc((size_t)E_*HID_*2);
  _Float16* WcH    = (_Float16*)alloc((size_t)A_*E_*2);
  _Float16* Wm1H   = (_Float16*)alloc((size_t)HID_*A_*E_*2);
  _Float16* Wm2H   = (_Float16*)alloc((size_t)E_*HID_*2);
  float*    gi0    = (float*)   alloc((size_t)N_*3*HID_*4);
  _Float16* h0h    = (_Float16*)alloc((size_t)N_*HID_*2);
  float*    X      = (float*)   alloc((size_t)N_*HP_*E_*4);
  float*    logits = (float*)   alloc((size_t)N_*A_*4);
  int*      Ibuf   = (int*)     alloc((size_t)N_*4);
  int*      pa     = (int*)     alloc((size_t)N_*HP_*4);
  float*    xcur   = (float*)   alloc((size_t)N_*E_*4);
  float*    xaf    = (float*)   alloc((size_t)N_*E_*4);
  _Float16* xah    = (_Float16*)alloc((size_t)N_*E_*2);
  _Float16* flat   = (_Float16*)alloc((size_t)512*A_*E_*2);   // time chunk; plan uses first 64 rows
  _Float16* mhh    = (_Float16*)alloc((size_t)512*HID_*2);
  float*    vals   = (float*)   alloc((size_t)N_*A_*4);
  float*    xnext  = (float*)   alloc((size_t)N_*E_*4);

  // weight -> f16 conversions (per call: no cached state allowed)
  k_convert<<<(3*HID_*E_+255)/256, 256, 0, stream>>>(Wih, WihH, 3*HID_*E_);
  k_convert<<<(E_*HID_+255)/256,   256, 0, stream>>>(We2, We2H, E_*HID_);
  k_convert<<<(A_*E_+255)/256,     256, 0, stream>>>(Wc,  WcH,  A_*E_);
  k_convert<<<(HID_*A_*E_+255)/256,256, 0, stream>>>(Wm1, Wm1H, HID_*A_*E_);
  k_convert<<<(E_*HID_+255)/256,   256, 0, stream>>>(Wm2, Wm2H, E_*HID_);

  k_embed<<<(T_*N_*E_+255)/256, 256, 0, stream>>>(obs, emb1, emb, embh);
  k_init <<<512, 256, 0, stream>>>(X, logits, Ibuf, pa);

  // GRU (only step 0 is live): gi0 = emb[0] @ Wih^T + bih ; then gates; then X0
  k_gemm<<<dim3(48, 4), 32, 0, stream>>>(embh, WihH, bih, N_, 3*HID_, E_, 3*HID_,
                                         gi0, nullptr, nullptr, 0);
  k_gru0<<<(N_*HID_+255)/256, 256, 0, stream>>>(gi0, bhh, h0h);
  k_gemm<<<dim3(4, 4), 32, 0, stream>>>(h0h, We2H, be2, N_, E_, HID_, HP_*E_,
                                        X, nullptr, nullptr, 0);   // writes X[:,0,:]

  // planning scan (S sequential steps)
  for (int s = 0; s < S_; ++s) {
    k_gather<<<(N_*E_+255)/256, 256, 0, stream>>>(X, Ibuf, xcur, xaf, xah);
    k_flat_plan<<<(N_*A_*E_+255)/256, 256, 0, stream>>>(xcur, eopt, P + s*N_, flat);
    k_gemm<<<dim3(1, 4),  32, 0, stream>>>(xah,  WcH,  bc,  N_, A_,   E_,    A_,
                                           vals, nullptr, nullptr, 0);
    k_gemm<<<dim3(16, 4), 32, 0, stream>>>(flat, Wm1H, bm1, N_, HID_, A_*E_, HID_,
                                           nullptr, mhh, nullptr, 1);
    k_gemm<<<dim3(4, 4),  32, 0, stream>>>(mhh,  Wm2H, bm2, N_, E_,   HID_,  E_,
                                           xnext, nullptr, nullptr, 0);
    k_update<<<1, 64, 0, stream>>>(P + s*N_, xaf, Ws, bs, vals, xnext,
                                   logits, Ibuf, pa, X, out + SL_OFF + (size_t)s*N_*A_);
  }

  // time-loop model loss in 4 chunks of 512 rows; loss fused into GEMM epilogue
  for (int c = 0; c < 4; ++c) {
    k_flat_time<<<(512*A_*E_+255)/256, 256, 0, stream>>>(emb, eopt, P, flat, c);
    k_gemm<<<dim3(16, 32), 32, 0, stream>>>(flat, Wm1H, bm1, 512, HID_, A_*E_, HID_,
                                            nullptr, mhh, nullptr, 1);
    k_gemm<<<dim3(4, 32),  32, 0, stream>>>(mhh,  Wm2H, bm2, 512, E_,   HID_,  E_,
                                            out + MO_OFF + (size_t)c*512*E_, nullptr,
                                            emb + (size_t)c*512*E_, 0);
  }

  k_embed_loss<<<T_*N_, 32, 0, stream>>>(emb, X, Ws, bs, out + EL_OFF);
  k_finalize<<<(N_*HP_*E_+255)/256, 256, 0, stream>>>(X, pa, P, out);
}
